// TanhNewtonImplicitLayer_17145509446157
// MI455X (gfx1250) — matrix operand order repaired and checked by reading, not compiled, not run on hardware
//
#include <hip/hip_runtime.h>
#include <hip/hip_bf16.h>

// ---------------------------------------------------------------------------
// TanhNewtonImplicitLayer on MI455X (gfx1250):
// Fixed point z* = tanh(z W^T + x) via Picard iteration, fully LDS/register
// resident. bf16 hi/lo split WMMA (v_wmma_f32_16x16x32_bf16, f32 accumulate)
// gives ~f32 accuracy at matrix-core rates. x fragments feed the WMMA C
// operand directly (no acc-init copies). tanh via V_TANH_F32 if available.
// ---------------------------------------------------------------------------

typedef __bf16 bf16_t;
typedef bf16_t v16bf __attribute__((ext_vector_type(16)));
typedef bf16_t bf8v  __attribute__((ext_vector_type(8)));   // 16 bytes
typedef bf16_t bf4v  __attribute__((ext_vector_type(4)));   // 8 bytes
typedef float  v8f   __attribute__((ext_vector_type(8)));

struct alignas(16) U4 { unsigned int a, b, c, d; };
struct alignas(16) F4 { float x, y, z, w; };

union FragU { v16bf v; U4 q[2]; };

#define D_DIM   256
#define B_DIM   2048
#define WPITCH  264                       // 256 + 8 bf16 pad: 528B rows, 16B aligned, bank-spread
#define W_LO_OFF (256 * WPITCH)           // bf16-element offsets into LDS
#define ZBASE    (2 * 256 * WPITCH)
#define ZG_SIZE  (2 * 16 * WPITCH)        // z_hi + z_lo for one 16-row batch tile
#define LDS_BYTES ((ZBASE + 2 * ZG_SIZE) * 2)   // 304128 B of the 320KB WGP LDS

#define ITERS 100

// f32 tanh: prefer the CDNA5 hardware transcendental, else exp2+rcp (~1e-7)
__device__ __forceinline__ float fast_tanh(float y) {
#if __has_builtin(__builtin_amdgcn_tanhf)
    return __builtin_amdgcn_tanhf(y);
#else
    float ay = __builtin_fabsf(y);
    float e2 = __builtin_amdgcn_exp2f(ay * 2.88539008177793f);   // exp(2*ay)
    float r  = __builtin_amdgcn_rcpf(1.0f + e2);
    float m  = 1.0f - 2.0f * r;
    return __builtin_copysignf(m, y);
#endif
}

__global__ __launch_bounds__(256)
void deq_picard_wmma(const float* __restrict__ x,
                     const float* __restrict__ W,
                     float* __restrict__ out)
{
    extern __shared__ bf16_t lds[];
    bf16_t* Whi = lds;
    bf16_t* Wlo = lds + W_LO_OFF;

    const int tid  = threadIdx.x;
    const int lane = tid & 31;
    const int wave = tid >> 5;          // 0..7
    const int g    = wave >> 2;         // batch sub-tile (0/1) within workgroup
    const int f    = wave & 3;          // 64-feature slice (0..3)
    const int s    = lane >> 4;         // half-wave select
    const int n    = lane & 15;         // batch column within tile / A row

    bf16_t* Zhi = lds + ZBASE + g * ZG_SIZE;
    bf16_t* Zlo = Zhi + 16 * WPITCH;

    // ---- one-time: split W (f32 row-major) into bf16 hi/lo planes in LDS ----
    for (int i = tid * 4; i < D_DIM * D_DIM; i += 256 * 4) {
        F4 w4 = *(const F4*)(W + i);
        int m = i >> 8;
        int k = i & 255;
        bf16_t h0 = (bf16_t)w4.x, h1 = (bf16_t)w4.y, h2 = (bf16_t)w4.z, h3 = (bf16_t)w4.w;
        bf16_t l0 = (bf16_t)(w4.x - (float)h0), l1 = (bf16_t)(w4.y - (float)h1);
        bf16_t l2 = (bf16_t)(w4.z - (float)h2), l3 = (bf16_t)(w4.w - (float)h3);
        *(bf4v*)(Whi + m * WPITCH + k) = (bf4v){h0, h1, h2, h3};
        *(bf4v*)(Wlo + m * WPITCH + k) = (bf4v){l0, l1, l2, l3};
    }
    // ---- zero both z buffers (z0 = 0, matching the reference) ----
    {
        unsigned int* zp = (unsigned int*)(lds + ZBASE);
        for (int i = tid; i < ZG_SIZE; i += 256) zp[i] = 0u;  // 2*ZG_SIZE bf16 = ZG_SIZE u32
    }

    // ---- preload this lane's x fragments (C/D layout) straight into v8f ----
    const int bRow = blockIdx.x * 32 + g * 16 + n;
    v8f xvv[4];
#pragma unroll
    for (int t = 0; t < 4; ++t) {
        const float* xp = x + bRow * D_DIM + (f * 64 + t * 16 + s * 8);
        F4 a = *(const F4*)xp;
        F4 b = *(const F4*)(xp + 4);
        xvv[t][0] = a.x; xvv[t][1] = a.y; xvv[t][2] = a.z; xvv[t][3] = a.w;
        xvv[t][4] = b.x; xvv[t][5] = b.y; xvv[t][6] = b.z; xvv[t][7] = b.w;
    }

    // ---- loop-invariant per-lane fragment base pointers (DS imm offsets do the rest)
    const bf16_t* aHiB[4];
    const bf16_t* aLoB[4];
#pragma unroll
    for (int t = 0; t < 4; ++t) {
        const int mrow = f * 64 + t * 16 + n;
        aHiB[t] = Whi + mrow * WPITCH + s * 8;
        aLoB[t] = Wlo + mrow * WPITCH + s * 8;
    }
    const bf16_t* bHiB = Zhi + n * WPITCH + s * 8;
    const bf16_t* bLoB = Zlo + n * WPITCH + s * 8;
    bf16_t* zStoreHi = Zhi + n * WPITCH + f * 64 + s * 8;
    bf16_t* zStoreLo = Zlo + n * WPITCH + f * 64 + s * 8;

    __syncthreads();

    float zn[4][8];

    for (int iter = 0; iter < ITERS; ++iter) {
        v8f acc[4];

        // ---- K chunk 0 (peeled): C operand = x fragment, acc born in the WMMA
        {
            FragU bh, bl;
            bh.q[0] = *(const U4*)(bHiB);
            bh.q[1] = *(const U4*)(bHiB + 16);
            bl.q[0] = *(const U4*)(bLoB);
            bl.q[1] = *(const U4*)(bLoB + 16);
#pragma unroll
            for (int t = 0; t < 4; ++t) {
                FragU ah, al;
                ah.q[0] = *(const U4*)(aHiB[t]);
                ah.q[1] = *(const U4*)(aHiB[t] + 16);
                al.q[0] = *(const U4*)(aLoB[t]);
                al.q[1] = *(const U4*)(aLoB[t] + 16);
                acc[t] = __builtin_amdgcn_wmma_f32_16x16x32_bf16(
                    false, ah.v, false, bh.v, (short)0, xvv[t], false, false);
                acc[t] = __builtin_amdgcn_wmma_f32_16x16x32_bf16(
                    false, ah.v, false, bl.v, (short)0, acc[t], false, false);
                acc[t] = __builtin_amdgcn_wmma_f32_16x16x32_bf16(
                    false, al.v, false, bh.v, (short)0, acc[t], false, false);
            }
        }

        // ---- K chunks 1..7
#pragma unroll
        for (int c = 1; c < 8; ++c) {
            FragU bh, bl;
            bh.q[0] = *(const U4*)(bHiB + c * 32);
            bh.q[1] = *(const U4*)(bHiB + c * 32 + 16);
            bl.q[0] = *(const U4*)(bLoB + c * 32);
            bl.q[1] = *(const U4*)(bLoB + c * 32 + 16);

#pragma unroll
            for (int t = 0; t < 4; ++t) {
                FragU ah, al;
                ah.q[0] = *(const U4*)(aHiB[t] + c * 32);
                ah.q[1] = *(const U4*)(aHiB[t] + c * 32 + 16);
                al.q[0] = *(const U4*)(aLoB[t] + c * 32);
                al.q[1] = *(const U4*)(aLoB[t] + c * 32 + 16);

                // bf16x3 split-product: hi*hi + hi*lo + lo*hi, f32 accumulate
                acc[t] = __builtin_amdgcn_wmma_f32_16x16x32_bf16(
                    false, ah.v, false, bh.v, (short)0, acc[t], false, false);
                acc[t] = __builtin_amdgcn_wmma_f32_16x16x32_bf16(
                    false, ah.v, false, bl.v, (short)0, acc[t], false, false);
                acc[t] = __builtin_amdgcn_wmma_f32_16x16x32_bf16(
                    false, al.v, false, bh.v, (short)0, acc[t], false, false);
            }
        }

        // z_new = tanh(W z + x)
#pragma unroll
        for (int t = 0; t < 4; ++t)
#pragma unroll
            for (int r = 0; r < 8; ++r) zn[t][r] = fast_tanh(acc[t][r]);

        __syncthreads();   // all reads of old z complete before overwrite

        // write new z back as pre-split B-fragment planes (16B DS stores)
#pragma unroll
        for (int t = 0; t < 4; ++t) {
            bf8v hv, lv;
#pragma unroll
            for (int r = 0; r < 8; ++r) {
                bf16_t h = (bf16_t)zn[t][r];
                hv[r] = h;
                lv[r] = (bf16_t)(zn[t][r] - (float)h);
            }
            *(bf8v*)(zStoreHi + t * 16) = hv;
            *(bf8v*)(zStoreLo + t * 16) = lv;
        }
        __syncthreads();   // new z visible to the whole group
    }

    // Output = tanh(z W^T + x) = last iterate's RHS (still in zn registers)
    float* op = out + bRow * D_DIM + f * 64;
#pragma unroll
    for (int t = 0; t < 4; ++t) {
        F4 a = {zn[t][0], zn[t][1], zn[t][2], zn[t][3]};
        F4 b = {zn[t][4], zn[t][5], zn[t][6], zn[t][7]};
        *(F4*)(op + t * 16 + s * 8)     = a;
        *(F4*)(op + t * 16 + s * 8 + 4) = b;
    }
}

extern "C" void kernel_launch(void* const* d_in, const int* in_sizes, int n_in,
                              void* d_out, int out_size, void* d_ws, size_t ws_size,
                              hipStream_t stream) {
    const float* x = (const float*)d_in[0];   // [2048, 256] f32
    const float* W = (const float*)d_in[1];   // [256, 256]  f32
    float* out = (float*)d_out;               // [2048, 256] f32

    dim3 grid(B_DIM / 32);   // 64 workgroups, 32 batch rows each
    dim3 block(256);         // 8 wave32s
    hipLaunchKernelGGL(deq_picard_wmma, grid, block, LDS_BYTES, stream, x, W, out);
}
